// DecoderWithAttention_89154931130660
// MI455X (gfx1250) — matrix operand order repaired and checked by reading, not compiled, not run on hardware
//
#include <hip/hip_runtime.h>
#include <cmath>

// ---------------------------------------------------------------------------
// Types for CDNA5 WMMA (wave32): D(16x16 f32) = A(16x32 bf16) * B(32x16 bf16) + C
// ---------------------------------------------------------------------------
typedef __attribute__((ext_vector_type(16))) __bf16 v16bf;
typedef __attribute__((ext_vector_type(8)))  __bf16 v8bf;
typedef __attribute__((ext_vector_type(4)))  __bf16 v4bf;
typedef __attribute__((ext_vector_type(8)))  float  v8f;

// async-copy builtin operand types: pointer-to-int4-vector with explicit AS
typedef int v4i __attribute__((vector_size(16)));
typedef __attribute__((address_space(1))) v4i as1_v4i;
typedef __attribute__((address_space(3))) v4i as3_v4i;
typedef __attribute__((address_space(3))) void lds_void_t;

#define NV 50000
#define NE 256
#define NH 512
#define NB 64
#define NT 2048

__device__ __forceinline__ v8f wmma_bf16(v16bf a, v16bf b, v8f c) {
  return __builtin_amdgcn_wmma_f32_16x16x32_bf16(false, a, false, b, (short)0, c,
                                                 false, false);
}

__device__ __forceinline__ v4bf cvt4(float4 x) {
  v4bf y;
  y[0] = (__bf16)x.x; y[1] = (__bf16)x.y; y[2] = (__bf16)x.z; y[3] = (__bf16)x.w;
  return y;
}

// ---- CDNA5 async global->LDS copy (ASYNCcnt path) -------------------------
__device__ __forceinline__ void async_copy_16B(const __bf16* gsrc, __bf16* ldst) {
#if __has_builtin(__builtin_amdgcn_global_load_async_to_lds_b128)
  __builtin_amdgcn_global_load_async_to_lds_b128((as1_v4i*)gsrc, (as3_v4i*)ldst,
                                                 0, 0);
#else
  unsigned laddr = (unsigned)(size_t)(lds_void_t*)ldst;
  unsigned long long gaddr = (unsigned long long)(size_t)gsrc;
  asm volatile("global_load_async_to_lds_b128 %0, %1, off"
               :: "v"(laddr), "v"(gaddr) : "memory");
#endif
}

__device__ __forceinline__ void wait_async0() {
#if __has_builtin(__builtin_amdgcn_s_wait_asynccnt)
  __builtin_amdgcn_s_wait_asynccnt(0);
#else
  asm volatile("s_wait_asynccnt 0x0" ::: "memory");
#endif
}

// A-fragment (16x32 bf16) from an LDS row (row-major, contiguous k).
// lane L: row = L&15 (caller bakes into `row`); k = (L>>4)*16 + j
__device__ __forceinline__ v16bf lds_a_frag(const __bf16* row, int ks, int lane) {
  const int kb = ks * 32 + ((lane >> 4) << 4);
  v8bf c0 = *(const v8bf*)(row + kb);
  v8bf c1 = *(const v8bf*)(row + kb + 8);
  v16bf a;
#pragma unroll
  for (int j = 0; j < 8; ++j) { a[j] = c0[j]; a[j + 8] = c1[j]; }
  return a;
}

// B-fragment (32x16 bf16) from LDS tile stored n-major: bt[n*ldk + k].
// lane L: n = L&15; k = (L>>4)*16 + j
__device__ __forceinline__ v16bf lds_b_frag(const __bf16* bt, int ldk, int ks, int lane) {
  const int n = lane & 15;
  const int k = ks * 32 + ((lane >> 4) << 4);
  const v8bf* p = (const v8bf*)(bt + n * ldk + k);
  v8bf c0 = p[0], c1 = p[1];
  v16bf b;
#pragma unroll
  for (int j = 0; j < 8; ++j) { b[j] = c0[j]; b[j + 8] = c1[j]; }
  return b;
}

// B-fragment directly from global fp32 weight matrix W (rows = n, contiguous k).
__device__ __forceinline__ v16bf glb_b_frag(const float* W, int ldk, int n0, int kg,
                                            int lane) {
  const int n = n0 + (lane & 15);
  const int k = kg + ((lane >> 4) << 4);
  const float4* p = (const float4*)(W + (size_t)n * ldk + k);
  v16bf b;
#pragma unroll
  for (int q = 0; q < 4; ++q) {
    float4 f = p[q];
    b[q * 4 + 0] = (__bf16)f.x; b[q * 4 + 1] = (__bf16)f.y;
    b[q * 4 + 2] = (__bf16)f.z; b[q * 4 + 3] = (__bf16)f.w;
  }
  return b;
}

// A-fragment directly from a global fp32 row.
__device__ __forceinline__ v16bf glb_a_frag(const float* row, int ks, int lane) {
  const int kb = ks * 32 + ((lane >> 4) << 4);
  const float4* p0 = (const float4*)(row + kb);
  const float4* p1 = (const float4*)(row + kb + 8);
  v16bf a;
  float4 f;
  f = p0[0]; a[0]=(__bf16)f.x; a[1]=(__bf16)f.y; a[2]=(__bf16)f.z; a[3]=(__bf16)f.w;
  f = p0[1]; a[4]=(__bf16)f.x; a[5]=(__bf16)f.y; a[6]=(__bf16)f.z; a[7]=(__bf16)f.w;
  f = p1[0]; a[8]=(__bf16)f.x; a[9]=(__bf16)f.y; a[10]=(__bf16)f.z; a[11]=(__bf16)f.w;
  f = p1[1]; a[12]=(__bf16)f.x; a[13]=(__bf16)f.y; a[14]=(__bf16)f.z; a[15]=(__bf16)f.w;
  return a;
}

__device__ __forceinline__ float sigm(float x) { return 1.0f / (1.0f + __expf(-x)); }

// ---------------------------------------------------------------------------
// Kernel 0: pre-convert W_enc (H x H fp32) -> bf16 in workspace (same layout:
// row = h_out (= GEMM n), col = h_in (= GEMM k)). 1 MB read, 0.5 MB write.
// Enables async (no-convert) B-slab staging in k_attn_scores.
// ---------------------------------------------------------------------------
__global__ __launch_bounds__(256) void k_cvt_wenc(const float* __restrict__ W,
                                                  __bf16* __restrict__ Wb) {
  const int i = blockIdx.x * 256 + threadIdx.x;  // float4 index, NH*NH/4 total
  float4 x = ((const float4*)W)[i];
  *(v4bf*)&Wb[(size_t)i * 4] = cvt4(x);
}

// ---------------------------------------------------------------------------
// Kernel 1: dec_proj = h @ W_dec.T   (64 x 512 x 512)
// grid 32 (N tiles), block 128 (4 waves = 4 M tiles)
// ---------------------------------------------------------------------------
__global__ __launch_bounds__(128) void k_dec_proj(const float* __restrict__ h,
                                                  const float* __restrict__ Wdec,
                                                  float* __restrict__ dproj) {
  const int lane = threadIdx.x & 31;
  const int mt = threadIdx.x >> 5;
  const int n0 = blockIdx.x * 16;
  const float* arow = h + (size_t)(mt * 16 + (lane & 15)) * NH;
  v8f acc = {};
  for (int ks = 0; ks < 16; ++ks) {
    v16bf a = glb_a_frag(arow, ks, lane);
    v16bf b = glb_b_frag(Wdec, NH, n0, ks * 32, lane);
    acc = wmma_bf16(a, b, acc);
  }
  const int n = n0 + (lane & 15);
  const int mb = mt * 16 + ((lane >> 4) << 3);
#pragma unroll
  for (int v = 0; v < 8; ++v) dproj[(mb + v) * NH + n] = acc[v];
}

// ---------------------------------------------------------------------------
// Kernel 2: scores[b,t] = v_attn . tanh(enc[b] @ W_enc.T + dec_proj[b])
// grid (T/128, B), block 256 (8 waves x 16 rows of t). Energy never hits HBM.
// dyn LDS: 128x512 bf16 A (reused over 32 N tiles)
//        + double-buffered 16x512 bf16 B slab filled by ASYNC global->LDS.
// ---------------------------------------------------------------------------
__global__ __launch_bounds__(256) void k_attn_scores(const float* __restrict__ enc,
                                                     const __bf16* __restrict__ Wencb,
                                                     const float* __restrict__ dproj,
                                                     const float* __restrict__ vattn,
                                                     float* __restrict__ scores) {
  extern __shared__ __bf16 smem1[];
  __bf16* sA = smem1;              // [128][512]
  __bf16* sB = smem1 + 128 * 512;  // [2][16][512] n-major, async-filled
  const int b = blockIdx.y;
  const int t0 = blockIdx.x * 128;
  const int tid = threadIdx.x;
  const int lane = tid & 31;
  const int wave = tid >> 5;

  // kick off async fill of B slab for tile 0 (each thread copies 64 B = 4x b128)
  {
    const __bf16* gsrc = Wencb + (size_t)tid * 32;  // n0 = 0
    __bf16* ldst = sB + tid * 32;
#pragma unroll
    for (int q = 0; q < 4; ++q) async_copy_16B(gsrc + q * 8, ldst + q * 8);
  }

  // stage + convert the A block (overlaps with the async B fill)
  const float* encB = enc + ((size_t)b * NT + t0) * NH;
#pragma unroll 4
  for (int i = tid; i < 128 * 512 / 4; i += 256) {
    float4 x = ((const float4*)encB)[i];
    *(v4bf*)&sA[i * 4] = cvt4(x);
  }

  const __bf16* arow = sA + (size_t)(wave * 16 + (lane & 15)) * 512;
  float accS[8] = {0.f, 0.f, 0.f, 0.f, 0.f, 0.f, 0.f, 0.f};

  for (int nt = 0; nt < 32; ++nt) {
    wait_async0();       // my async copies done
    __syncthreads();     // everyone's copies + A staging visible
    if (nt + 1 < 32) {   // prefetch next slab into the other buffer
      const __bf16* gsrc = Wencb + (size_t)(nt + 1) * 16 * 512 + tid * 32;
      __bf16* ldst = sB + ((nt + 1) & 1) * 8192 + tid * 32;
#pragma unroll
      for (int q = 0; q < 4; ++q) async_copy_16B(gsrc + q * 8, ldst + q * 8);
    }
    const __bf16* bt = sB + (nt & 1) * 8192;

    v8f acc = {};
#pragma unroll 4
    for (int ks = 0; ks < 16; ++ks) {
      v16bf a = lds_a_frag(arow, ks, lane);
      v16bf bb = lds_b_frag(bt, 512, ks, lane);
      acc = wmma_bf16(a, bb, acc);
    }
    const int n = nt * 16 + (lane & 15);
    const float dp = dproj[b * NH + n];
    const float va = vattn[n];
#pragma unroll
    for (int v = 0; v < 8; ++v) accS[v] += tanhf(acc[v] + dp) * va;
    // no trailing barrier: next iteration's wait+barrier protects buffer reuse
  }

#pragma unroll
  for (int v = 0; v < 8; ++v) {
    accS[v] += __shfl_xor(accS[v], 1, 32);
    accS[v] += __shfl_xor(accS[v], 2, 32);
    accS[v] += __shfl_xor(accS[v], 4, 32);
    accS[v] += __shfl_xor(accS[v], 8, 32);
  }
  if ((lane & 15) == 0) {
    const int mb = wave * 16 + ((lane >> 4) << 3);
#pragma unroll
    for (int v = 0; v < 8; ++v) scores[b * NT + t0 + mb + v] = accS[v];
  }
}

// ---------------------------------------------------------------------------
// Kernel 3: softmax over T per batch. grid 64, block 256 (8 elems/thread)
// ---------------------------------------------------------------------------
__global__ __launch_bounds__(256) void k_softmax(const float* __restrict__ scores,
                                                 float* __restrict__ attn) {
  __shared__ float red[256];
  const int b = blockIdx.x;
  const int tid = threadIdx.x;
  const float* s = scores + b * NT;
  float v[8];
  float mx = -3.402823466e+38f;
#pragma unroll
  for (int i = 0; i < 8; ++i) { v[i] = s[tid + i * 256]; mx = fmaxf(mx, v[i]); }
  red[tid] = mx;
  __syncthreads();
  for (int o = 128; o > 0; o >>= 1) {
    if (tid < o) red[tid] = fmaxf(red[tid], red[tid + o]);
    __syncthreads();
  }
  mx = red[0];
  __syncthreads();
  float sum = 0.f;
#pragma unroll
  for (int i = 0; i < 8; ++i) { v[i] = __expf(v[i] - mx); sum += v[i]; }
  red[tid] = sum;
  __syncthreads();
  for (int o = 128; o > 0; o >>= 1) {
    if (tid < o) red[tid] += red[tid + o];
    __syncthreads();
  }
  const float inv = 1.0f / red[0];
#pragma unroll
  for (int i = 0; i < 8; ++i) attn[b * NT + tid + i * 256] = v[i] * inv;
}

// ---------------------------------------------------------------------------
// Kernel 4: context[b,h] = sum_t attn[b,t] * enc[b,t,h]  (bandwidth bound)
// grid (H/256, B), block 256 = 64 lanes x float4 cols x 4 t-slices
// ---------------------------------------------------------------------------
__global__ __launch_bounds__(256) void k_context(const float* __restrict__ enc,
                                                 const float* __restrict__ attn,
                                                 float* __restrict__ ctx) {
  __shared__ float4 part[256];
  const int b = blockIdx.y;
  const int tid = threadIdx.x;
  const int hc = blockIdx.x * 256 + (tid & 63) * 4;  // float4 column base
  const int slice = tid >> 6;
  const float* e = enc + (size_t)b * NT * NH;
  const float* a = attn + b * NT;
  float4 acc = {0.f, 0.f, 0.f, 0.f};
  for (int t = slice; t < NT; t += 4) {
    const float w = a[t];
    float4 x = *(const float4*)(e + (size_t)t * NH + hc);
    acc.x += w * x.x; acc.y += w * x.y; acc.z += w * x.z; acc.w += w * x.w;
  }
  part[tid] = acc;
  __syncthreads();
  if (slice == 0) {
    float4 r = part[tid], p1 = part[tid + 64], p2 = part[tid + 128],
           p3 = part[tid + 192];
    r.x += p1.x + p2.x + p3.x;
    r.y += p1.y + p2.y + p3.y;
    r.z += p1.z + p2.z + p3.z;
    r.w += p1.w + p2.w + p3.w;
    *(float4*)(ctx + b * NH + hc) = r;
  }
}

// ---------------------------------------------------------------------------
// Kernel 5: gates = [emb[tok] | ctx | h] @ [W_ih ; W_hh]^T + b_ih + b_hh
// M=64 (4 tiles, grid.y), N=2048 (128 tiles: grid.x=16 x 8 waves), K=1280
// ---------------------------------------------------------------------------
__global__ __launch_bounds__(256) void k_gates(
    const int* __restrict__ tok, const float* __restrict__ emb,
    const float* __restrict__ ctx, const float* __restrict__ hprev,
    const float* __restrict__ Wih, const float* __restrict__ Whh,
    const float* __restrict__ bih, const float* __restrict__ bhh,
    float* __restrict__ gates) {
  __shared__ __bf16 sA[16 * 1280];  // 40 KB
  const int mt = blockIdx.y;
  const int tid = threadIdx.x;
  const int lane = tid & 31;
  const int wave = tid >> 5;
  for (int i = tid; i < 16 * 1280 / 4; i += 256) {
    const int idx = i * 4;
    const int r = idx / 1280;
    const int k = idx % 1280;
    const int m = mt * 16 + r;
    const float* src;
    if (k < 256)      src = emb + (size_t)tok[m] * NE + k;
    else if (k < 768) src = ctx + m * NH + (k - 256);
    else              src = hprev + m * NH + (k - 768);
    *(v4bf*)&sA[idx] = cvt4(*(const float4*)src);
  }
  __syncthreads();
  const int n0 = (blockIdx.x * 8 + wave) * 16;
  const __bf16* arow = sA + (lane & 15) * 1280;
  v8f acc = {};
  for (int ks = 0; ks < 40; ++ks) {
    v16bf a = lds_a_frag(arow, ks, lane);
    const int kg = ks * 32;
    v16bf b = (kg < 768) ? glb_b_frag(Wih, 768, n0, kg, lane)
                         : glb_b_frag(Whh, 512, n0, kg - 768, lane);
    acc = wmma_bf16(a, b, acc);
  }
  const int n = n0 + (lane & 15);
  const float bias = bih[n] + bhh[n];
  const int mb = mt * 16 + ((lane >> 4) << 3);
#pragma unroll
  for (int v = 0; v < 8; ++v) gates[(mb + v) * 2048 + n] = acc[v] + bias;
}

// ---------------------------------------------------------------------------
// Kernel 6: LSTM pointwise (gate order i, f, g, o)
// ---------------------------------------------------------------------------
__global__ __launch_bounds__(256) void k_lstm(const float* __restrict__ gates,
                                              const float* __restrict__ cprev,
                                              float* __restrict__ hout,
                                              float* __restrict__ cout) {
  const int idx = blockIdx.x * 256 + threadIdx.x;  // 0 .. 64*512-1
  const int b = idx >> 9;
  const int hc = idx & 511;
  const float* g = gates + b * 2048;
  const float ig = sigm(g[hc]);
  const float fg = sigm(g[512 + hc]);
  const float gg = tanhf(g[1024 + hc]);
  const float og = sigm(g[1536 + hc]);
  const float c = fg * cprev[idx] + ig * gg;
  cout[idx] = c;
  hout[idx] = og * tanhf(c);
}

// ---------------------------------------------------------------------------
// Kernel 7: logits = [h_new | ctx] @ W_out.T + b_out   (64 x 50000 x 1024)
// Memory bound on W_out (205 MB, read once). grid 782 x 4 waves; each wave
// owns a 16-col vocab tile and all 4 M tiles (4 WMMAs per B load).
// dyn LDS: 64x1024 bf16 A.
// ---------------------------------------------------------------------------
__global__ __launch_bounds__(128) void k_logits(const float* __restrict__ hnew,
                                                const float* __restrict__ ctx,
                                                const float* __restrict__ Wout,
                                                const float* __restrict__ bout,
                                                float* __restrict__ logits) {
  extern __shared__ __bf16 smem2[];  // [64][1024]
  const int tid = threadIdx.x;
  const int lane = tid & 31;
  const int wave = tid >> 5;
  for (int i = tid; i < 64 * 1024 / 4; i += 128) {
    const int idx = i * 4;
    const int m = idx >> 10;
    const int k = idx & 1023;
    const float* src = (k < 512) ? (hnew + m * 512 + k) : (ctx + m * 512 + (k - 512));
    *(v4bf*)&smem2[idx] = cvt4(*(const float4*)src);
  }
  __syncthreads();
  const int nt = blockIdx.x * 4 + wave;
  if (nt >= NV / 16) return;  // uniform per wave: EXEC stays all-ones for WMMA
  const int n0 = nt * 16;
  const __bf16* a0 = smem2 + (size_t)(0 + (lane & 15)) * 1024;
  const __bf16* a1 = smem2 + (size_t)(16 + (lane & 15)) * 1024;
  const __bf16* a2 = smem2 + (size_t)(32 + (lane & 15)) * 1024;
  const __bf16* a3 = smem2 + (size_t)(48 + (lane & 15)) * 1024;
  v8f acc0 = {}, acc1 = {}, acc2 = {}, acc3 = {};
  for (int ks = 0; ks < 32; ++ks) {
    v16bf b = glb_b_frag(Wout, 1024, n0, ks * 32, lane);
    if (ks + 1 < 32)  // stream-prefetch next 128B chunk of this lane's W_out row
      __builtin_prefetch(Wout + (size_t)(n0 + (lane & 15)) * 1024 + (ks + 1) * 32, 0, 1);
    acc0 = wmma_bf16(lds_a_frag(a0, ks, lane), b, acc0);
    acc1 = wmma_bf16(lds_a_frag(a1, ks, lane), b, acc1);
    acc2 = wmma_bf16(lds_a_frag(a2, ks, lane), b, acc2);
    acc3 = wmma_bf16(lds_a_frag(a3, ks, lane), b, acc3);
  }
  const int n = n0 + (lane & 15);
  const float bb = bout[n];
  const int mb = (lane >> 4) << 3;
#pragma unroll
  for (int v = 0; v < 8; ++v) {
    logits[(size_t)(0 + mb + v) * NV + n] = acc0[v] + bb;
    logits[(size_t)(16 + mb + v) * NV + n] = acc1[v] + bb;
    logits[(size_t)(32 + mb + v) * NV + n] = acc2[v] + bb;
    logits[(size_t)(48 + mb + v) * NV + n] = acc3[v] + bb;
  }
}

// ---------------------------------------------------------------------------
extern "C" void kernel_launch(void* const* d_in, const int* in_sizes, int n_in,
                              void* d_out, int out_size, void* d_ws, size_t ws_size,
                              hipStream_t stream) {
  const int*   tok    = (const int*)d_in[0];
  const float* hidden = (const float*)d_in[1];   // (1,B,H)
  const float* cell   = (const float*)d_in[2];   // (1,B,H)
  const float* enc    = (const float*)d_in[3];   // (B,T,H)
  const float* emb    = (const float*)d_in[4];   // (V,E)
  const float* Wenc   = (const float*)d_in[5];   // (H,H)
  const float* Wdec   = (const float*)d_in[6];   // (H,H)
  const float* vattn  = (const float*)d_in[7];   // (H,)
  const float* Wih    = (const float*)d_in[8];   // (4H, E+H)
  const float* Whh    = (const float*)d_in[9];   // (4H, H)
  const float* bih    = (const float*)d_in[10];
  const float* bhh    = (const float*)d_in[11];
  const float* Wout   = (const float*)d_in[12];  // (V, 2H)
  const float* bout   = (const float*)d_in[13];

  float* out = (float*)d_out;
  float* out_logits = out;                              // 64*50000
  float* out_h      = out + (size_t)NB * NV;            // 64*512
  float* out_c      = out_h + NB * NH;                  // 64*512
  float* out_attn   = out_c + NB * NH;                  // 64*2048

  float* ws     = (float*)d_ws;
  float* dproj  = ws;                     // 64*512
  float* scores = dproj + NB * NH;        // 64*2048
  float* ctx    = scores + NB * NT;       // 64*512
  float* gates  = ctx + NB * NH;          // 64*2048
  __bf16* wencb = (__bf16*)(gates + NB * 2048);  // 512*512 bf16 (16B-aligned)

  k_cvt_wenc<<<dim3(NH * NH / 4 / 256), dim3(256), 0, stream>>>(Wenc, wencb);

  k_dec_proj<<<dim3(NH / 16), dim3(128), 0, stream>>>(hidden, Wdec, dproj);

  const size_t lds1 = (size_t)(128 * 512 + 2 * 16 * 512) * sizeof(__bf16);  // 160 KB
  k_attn_scores<<<dim3(NT / 128, NB), dim3(256), lds1, stream>>>(enc, wencb, dproj,
                                                                 vattn, scores);

  k_softmax<<<dim3(NB), dim3(256), 0, stream>>>(scores, out_attn);

  k_context<<<dim3(NH / 256, NB), dim3(256), 0, stream>>>(enc, out_attn, ctx);

  k_gates<<<dim3(16, 4), dim3(256), 0, stream>>>(tok, emb, ctx, hidden, Wih, Whh,
                                                 bih, bhh, gates);

  k_lstm<<<dim3(NB * NH / 256), dim3(256), 0, stream>>>(gates, cell, out_h, out_c);

  const size_t lds2 = (size_t)(64 * 1024) * sizeof(__bf16);  // 128 KB
  k_logits<<<dim3((NV / 16 + 3) / 4), dim3(128), lds2, stream>>>(out_h, ctx, Wout,
                                                                 bout, out_logits);
}
